// CosmosUnpatch3D_30880814858317
// MI455X (gfx1250) — compile-verified, matrix operand order checked by reading, not run on hardware
//
#include <hip/hip_runtime.h>
#include <hip/hip_bf16.h>

// CosmosUnpatch3D: 2-level 3D inverse Haar DWT == constant 64x64 (+/-1) matrix
// applied to each 64-channel fiber. WMMA f16 GEMM, f32 accumulation.
//
//   Out[d, n] = sum_s H[d][s] * In[s, n]
//   d = dt*16 + dh*4 + dw (4x4x4 patch offset), s = s1*8 + s2 (ch = 24*s1+3*s2+c)
//   H[d][s] = (-1)^( popc(s1&d1) + popc(s2&d2) ),  all coefficients exactly +/-1.
//
// Bandwidth-bound: 233 MB moved -> ~10 us floor at 23.3 TB/s; the 4 GFLOP of
// f16 WMMA is negligible. A-operands are baked into a constexpr __constant__
// table in the exact 16-bit A-matrix lane layout (no VALU construction).

typedef __attribute__((ext_vector_type(16))) _Float16     v16h;
typedef __attribute__((ext_vector_type(8)))  float        v8f;
typedef __attribute__((ext_vector_type(4)))  float        v4f;
typedef __attribute__((ext_vector_type(8)))  unsigned int v8u;

#define THW      81920              // 5*128*128 sites per (b,c) slab
#define HW_IN    16384              // 128*128
#define OUT_HW   262144             // 512*512
#define OUT_T    17                 // 4*5 - 3 trimmed frames
#define NTILES_PER_SLAB 5120        // 81920/16
#define TPW      4                  // N-tiles per wave

// ---------------- compile-time Haar operand table ----------------
constexpr int hpar(int d, int s) {
  int dt = d >> 4, dh = (d >> 2) & 3, dw = d & 3;
  int d1 = ((dt >> 1) << 2) | ((dh >> 1) << 1) | (dw >> 1);
  int d2 = ((dt & 1)  << 2) | ((dh & 1)  << 1) | (dw & 1);
  int a = (s >> 3) & d1, b = (s & 7) & d2;
  int pc = (a & 1) + ((a >> 1) & 1) + ((a >> 2) & 1)
         + (b & 1) + ((b >> 1) & 1) + ((b >> 2) & 1);
  return pc & 1;
}

struct alignas(32) HTab { unsigned int w[4][2][32][8]; };

constexpr HTab make_htab() {
  HTab t{};
  for (int m = 0; m < 4; ++m)
    for (int kc = 0; kc < 2; ++kc)
      for (int l = 0; l < 32; ++l) {
        int hl = l >> 4, Mloc = l & 15;
        for (int v = 0; v < 8; ++v) {
          unsigned int dword = 0;
          for (int j = 0; j < 2; ++j) {
            // 16-bit A layout: lane half hl holds K = {0..7,16..23} + 8*hl,
            // two K values packed per dword.
            int K = ((v & 3) << 1) + j + 8 * hl + 16 * (v >> 2);
            int s = K + 32 * kc;
            int d = m * 16 + Mloc;
            unsigned int h16 = 0x3C00u | (hpar(d, s) ? 0x8000u : 0u); // +/-1.0h
            dword |= h16 << (16 * j);
          }
          t.w[m][kc][l][v] = dword;
        }
      }
  return t;
}

__device__ __constant__ HTab g_htab = make_htab();

__global__ __launch_bounds__(256) void haar_unpatch_wmma(
    const float* __restrict__ in, float* __restrict__ out) {
  const int lane = threadIdx.x & 31;
  const int wv   = threadIdx.x >> 5;
  const int Mloc = lane & 15;     // A row / B,C column within tile
  const int hl   = lane >> 4;     // lane half: selects K/M sub-block

  // ---- Load the 8 constant A operands (one 32B per-lane read each, WGP$-hot).
  v16h A[4][2];
  #pragma unroll
  for (int m = 0; m < 4; ++m)
    #pragma unroll
    for (int kc = 0; kc < 2; ++kc)
      A[m][kc] = __builtin_bit_cast(
          v16h, *(const v8u*)(&g_htab.w[m][kc][lane][0]));

  #pragma unroll
  for (int i = 0; i < TPW; ++i) {
    // 8 waves of a block take 8 consecutive N-tiles per iteration.
    int tau = __builtin_amdgcn_readfirstlane(
        (int)blockIdx.x * (8 * TPW) + i * 8 + wv);   // wave-uniform -> SGPR
    int slab = tau / NTILES_PER_SLAB;                // slab = b*3 + c
    int n0   = (tau - slab * NTILES_PER_SLAB) * 16;
    int b    = slab / 3;
    int cc   = slab - b * 3;

    // ---- B gather: 64 channel rows x 16 contiguous columns -> f16.
    // channel(e,kc) = 24*(hl + 2*(e>>3) + 4*kc) + 3*(e&7): fold hl into the
    // base pointer so every gather offset is a compile-time constant.
    const float* pin = in + (size_t)b * (192 * THW) + (size_t)cc * THW
                          + (size_t)hl * (24 * THW) + n0 + Mloc;
    v16h Bm[2];
    #pragma unroll
    for (int kc = 0; kc < 2; ++kc) {
      v16h bb{};
      #pragma unroll
      for (int e = 0; e < 16; ++e) {
        const int off = THW * (48 * (e >> 3) + 96 * kc + 3 * (e & 7));
        bb[e] = (_Float16)pin[off];
      }
      Bm[kc] = bb;
    }

    // ---- 8x v_wmma_f32_16x16x32_f16: the wave's full 64x16 output block.
    v8f Cacc[4];
    #pragma unroll
    for (int m = 0; m < 4; ++m) {
      v8f c0 = {};
      c0 = __builtin_amdgcn_wmma_f32_16x16x32_f16(
              false, A[m][0], false, Bm[0], (short)0, c0, false, false);
      c0 = __builtin_amdgcn_wmma_f32_16x16x32_f16(
              false, A[m][1], false, Bm[1], (short)0, c0, false, false);
      Cacc[m] = c0;
    }

    // ---- Scatter 4x4x4 patches. C layout: VGPR r, lane -> d = m*16+r+8*hl,
    // n = n0+Mloc. r=0..3 => dh=2*hl, dw=0..3 (contiguous float4 in w);
    // r=4..7 => dh=2*hl+1. Fully coalesced b128 stores.
    int t   = n0 / HW_IN;           // SGPR math (n0 is wave-uniform)
    int rem = n0 - t * HW_IN;
    int h   = rem >> 7;
    int w0  = rem & 127;
    float* pout = out + (size_t)slab * (OUT_T * OUT_HW);
    int wcol = 4 * (w0 + Mloc);
    #pragma unroll
    for (int m = 0; m < 4; ++m) {
      int ot = 4 * t + m - 3;       // temporal trim: drop first 3 frames
      if (ot < 0) continue;         // scalar branch (t, m wave-uniform)
      float* prow = pout + (size_t)ot * OUT_HW
                         + (size_t)(4 * h + 2 * hl) * 512 + wcol;
      v4f lo = { Cacc[m][0], Cacc[m][1], Cacc[m][2], Cacc[m][3] };
      v4f hi = { Cacc[m][4], Cacc[m][5], Cacc[m][6], Cacc[m][7] };
      *(v4f*)(prow)       = lo;     // row dh = 2*hl
      *(v4f*)(prow + 512) = hi;     // row dh = 2*hl + 1
    }
  }
}

extern "C" void kernel_launch(void* const* d_in, const int* in_sizes, int n_in,
                              void* d_out, int out_size, void* d_ws, size_t ws_size,
                              hipStream_t stream) {
  (void)in_sizes; (void)n_in; (void)out_size; (void)d_ws; (void)ws_size;
  const float* x = (const float*)d_in[0];
  float* out = (float*)d_out;
  // 30720 N-tiles total = 960 blocks * 8 waves * 4 tiles/wave
  dim3 grid(960), block(256);
  haar_unpatch_wmma<<<grid, block, 0, stream>>>(x, out);
}
